// NystromAttention_22849226015486
// MI455X (gfx1250) — compile-verified, matrix operand order checked by reading, not run on hardware
//
#include <hip/hip_runtime.h>

#define NHEAD 12
#define SEQ 4096
#define HD 64
#define BH 96
#define LMSZ 4096            // 64x64 per (b,h)
__constant__ const float kScale = 0.35355339059327373f; // 64^-0.25
#define NEGBIG 1000000000.0f

typedef __attribute__((ext_vector_type(16))) __bf16 v16bf;
typedef __attribute__((ext_vector_type(8)))  float  v8f;

__device__ __forceinline__ unsigned short f2bf(float x){
  union { float f; unsigned int u; } c; c.f = x;
  unsigned int r = c.u + 0x7FFFu + ((c.u >> 16) & 1u);
  return (unsigned short)(r >> 16);
}
__device__ __forceinline__ float bf2f(unsigned short h){
  union { unsigned int u; float f; } c; c.u = ((unsigned int)h) << 16; return c.f;
}
__device__ __forceinline__ unsigned int pk2(float a, float b){
  return (unsigned int)f2bf(a) | ((unsigned int)f2bf(b) << 16);
}

// Load a 16x32 bf16 fragment (A-layout; also B when source is Mat[n][k]-major).
__device__ __forceinline__ v16bf ld_frag(const unsigned short* base, int rc, int kBase, int ldm){
  int lane = threadIdx.x & 31;
  int g = lane >> 4, r = lane & 15;
  const unsigned short* p = base + (rc + r) * ldm + kBase + 8 * g;
  union { v16bf v; unsigned int u[8]; } f;
#pragma unroll
  for (int q = 0; q < 4; ++q){
    f.u[q]     = *(const unsigned int*)(p + 2 * q);        // K = 2q, 2q+1   (+8g)
    f.u[q + 4] = *(const unsigned int*)(p + 16 + 2 * q);   // K = 16+2q, ... (+8g)
  }
  return f.v;
}
__device__ __forceinline__ v8f wmma_bf16(v16bf a, v16bf b, v8f c){
  return __builtin_amdgcn_wmma_f32_16x16x32_bf16(false, a, false, b, (short)0, c, false, false);
}

// ---------------- Kernel A: landmarks (mean over 64-row segments) ----------------
// grid (96,8), block 128: task = (landmark l in 0..7, float4 column f4 in 0..15)
__global__ void lm_kernel(const float* __restrict__ Q, const float* __restrict__ K,
                          const float* __restrict__ mask,
                          float* __restrict__ wsQl, float* __restrict__ wsKl){
  int bh = blockIdx.x, b = bh / NHEAD;
  int l0 = blockIdx.y * 8;
  int t = threadIdx.x;
  int f4 = t & 15, l = t >> 4;                 // 128 threads
  __shared__ float sm[512];
  for (int i = t; i < 512; i += 128) sm[i] = mask[b * SEQ + l0 * 64 + i];
  __syncthreads();
  const float c = kScale * (1.0f / 64.0f);
  const float4* qp = (const float4*)(Q + ((size_t)bh * SEQ + (size_t)(l0 + l) * 64) * HD) + f4;
  const float4* kp = (const float4*)(K + ((size_t)bh * SEQ + (size_t)(l0 + l) * 64) * HD) + f4;
  float4 aq = {0.f,0.f,0.f,0.f}, ak = {0.f,0.f,0.f,0.f};
  for (int r = 0; r < 64; ++r){
    float m = sm[l * 64 + r];
    float4 qv = qp[r * 16];
    float4 kv = kp[r * 16];
    aq.x += qv.x * m; aq.y += qv.y * m; aq.z += qv.z * m; aq.w += qv.w * m;
    ak.x += kv.x * m; ak.y += kv.y * m; ak.z += kv.z * m; ak.w += kv.w * m;
  }
  float4 oq = {aq.x*c, aq.y*c, aq.z*c, aq.w*c};
  float4 ok = {ak.x*c, ak.y*c, ak.z*c, ak.w*c};
  ((float4*)(wsQl + bh * LMSZ + (l0 + l) * 64))[f4] = oq;
  ((float4*)(wsKl + bh * LMSZ + (l0 + l) * 64))[f4] = ok;
}

// ---------------- Kernel B: kernel_2 softmax + colsum max ----------------
__global__ void k2_kernel(const float* __restrict__ wsQl, const float* __restrict__ wsKl,
                          float* __restrict__ wsK2, float* __restrict__ wsCmax){
  int bh = blockIdx.x, t = threadIdx.x;      // 64 threads, thread t = row t
  __shared__ float sQ[64 * 65];
  __shared__ float sK[64 * 64];
  __shared__ float sS[64 * 65];
  __shared__ float scol[64];
  for (int i = t; i < 4096; i += 64){
    int r = i >> 6, c = i & 63;
    sQ[r * 65 + c] = wsQl[bh * LMSZ + i];
    sK[i]          = wsKl[bh * LMSZ + i];
  }
  __syncthreads();
  float row[64];
  float mx = -1e30f;
  for (int j = 0; j < 64; ++j){
    float s = 0.f;
    for (int k = 0; k < 64; ++k) s += sQ[t * 65 + k] * sK[j * 64 + k];
    row[j] = s; mx = fmaxf(mx, s);
  }
  float sum = 0.f;
  for (int j = 0; j < 64; ++j){ float e = __expf(row[j] - mx); row[j] = e; sum += e; }
  float inv = 1.0f / sum;
  for (int j = 0; j < 64; ++j){
    float v = row[j] * inv;
    sS[t * 65 + j] = v;
    wsK2[bh * LMSZ + t * 64 + j] = v;
  }
  __syncthreads();
  float cs = 0.f;
  for (int i = 0; i < 64; ++i) cs += sS[i * 65 + t];
  scol[t] = cs;
  __syncthreads();
  if (t == 0){
    float m = scol[0];
    for (int j = 1; j < 64; ++j) m = fmaxf(m, scol[j]);
    wsCmax[bh] = m;
  }
}

__global__ void gmax_kernel(const float* __restrict__ wsCmax, float* __restrict__ wsGmax){
  __shared__ float s[96];
  int t = threadIdx.x;
  if (t < 96) s[t] = wsCmax[t];
  __syncthreads();
  if (t == 0){
    float m = s[0];
    for (int i = 1; i < 96; ++i) m = fmaxf(m, s[i]);
    wsGmax[0] = m;
  }
}

// ---------------- Kernel C: streaming kernel_3 and T = kernel_3 @ V ----------------
// grid (96, 8); block 128 (4 waves). Split-K over s with f32 atomics into workspace.
__global__ void flash3_kernel(const float* __restrict__ K, const float* __restrict__ V,
                              const float* __restrict__ mask, const float* __restrict__ wsQl,
                              float* __restrict__ wsTacc, float* __restrict__ wsRow){
  int bh = blockIdx.x, b = bh / NHEAD, chunk = blockIdx.y;
  int t = threadIdx.x, w = t >> 5, lane = t & 31, g = lane >> 4, n = lane & 15;
  __shared__ __align__(16) unsigned short sQl[4096], sK[4096], sVt[4096], sP[4096];
  __shared__ float sM[64], sR[64];
  {
    const float4* Ql4 = (const float4*)(wsQl + bh * LMSZ);
    for (int j = t; j < 1024; j += 128){
      float4 v = Ql4[j];
      uint2 u; u.x = pk2(v.x, v.y); u.y = pk2(v.z, v.w);
      *(uint2*)(&sQl[j * 4]) = u;
    }
  }
  if (t < 64) sR[t] = 0.f;
  __syncthreads();
  v16bf aq0 = ld_frag(sQl, 16 * w, 0, 64);
  v16bf aq1 = ld_frag(sQl, 16 * w, 32, 64);
  v8f acc0 = {}, acc1 = {}, acc2 = {}, acc3 = {};
  const float* Kbase = K + (size_t)bh * SEQ * HD;
  const float* Vbase = V + (size_t)bh * SEQ * HD;
  for (int st = 0; st < 8; ++st){
    int s0 = chunk * 512 + st * 64;
    __syncthreads();                                    // prior tile fully consumed
    if (t < 64) sM[t] = mask[b * SEQ + s0 + t];
    __syncthreads();
    {
      const float4* K4 = (const float4*)(Kbase + (size_t)s0 * HD);
      const float4* V4 = (const float4*)(Vbase + (size_t)s0 * HD);
      for (int j = t; j < 1024; j += 128){
        int r = j >> 4, c4 = (j & 15) * 4;
        float ms = sM[r] * kScale;
        float4 kv = K4[j];
        uint2 u; u.x = pk2(kv.x * ms, kv.y * ms); u.y = pk2(kv.z * ms, kv.w * ms);
        *(uint2*)(&sK[r * 64 + c4]) = u;                // Ks masked+scaled, row-major
        float4 vv = V4[j];
        sVt[(c4 + 0) * 64 + r] = f2bf(vv.x);            // V transposed (B operand)
        sVt[(c4 + 1) * 64 + r] = f2bf(vv.y);
        sVt[(c4 + 2) * 64 + r] = f2bf(vv.z);
        sVt[(c4 + 3) * 64 + r] = f2bf(vv.w);
      }
      if (st + 1 < 8){                                  // one cacheline per lane = next tile
        const char* nk = (const char*)(Kbase + (size_t)(s0 + 64) * HD) + t * 128;
        const char* nv = (const char*)(Vbase + (size_t)(s0 + 64) * HD) + t * 128;
        __builtin_prefetch(nk, 0, 0);
        __builtin_prefetch(nv, 0, 0);
      }
    }
    __syncthreads();
    // logits = Ql @ Ks^T, then exp with mask bias -> sP (bf16, q-major)
#pragma unroll
    for (int ct = 0; ct < 4; ++ct){
      v8f l = {};
      l = wmma_bf16(aq0, ld_frag(sK, 16 * ct, 0, 64), l);
      l = wmma_bf16(aq1, ld_frag(sK, 16 * ct, 32, 64), l);
      float bias = NEGBIG * (1.0f - sM[16 * ct + n]);
#pragma unroll
      for (int j = 0; j < 8; ++j){
        int m = 16 * w + 8 * g + j;
        sP[m * 64 + 16 * ct + n] = f2bf(__expf(l[j] - bias));
      }
    }
    __syncthreads();
    if (t < 64){
      float rs = 0.f;
      for (int c = 0; c < 64; ++c) rs += bf2f(sP[t * 64 + c]);
      sR[t] += rs;
    }
    // T += P @ V  (A = P rows for this wave, B = Vt)
    v16bf ap0 = ld_frag(sP, 16 * w, 0, 64);
    v16bf ap1 = ld_frag(sP, 16 * w, 32, 64);
    acc0 = wmma_bf16(ap0, ld_frag(sVt, 0,  0, 64), acc0);
    acc0 = wmma_bf16(ap1, ld_frag(sVt, 0, 32, 64), acc0);
    acc1 = wmma_bf16(ap0, ld_frag(sVt, 16, 0, 64), acc1);
    acc1 = wmma_bf16(ap1, ld_frag(sVt, 16, 32, 64), acc1);
    acc2 = wmma_bf16(ap0, ld_frag(sVt, 32, 0, 64), acc2);
    acc2 = wmma_bf16(ap1, ld_frag(sVt, 32, 32, 64), acc2);
    acc3 = wmma_bf16(ap0, ld_frag(sVt, 48, 0, 64), acc3);
    acc3 = wmma_bf16(ap1, ld_frag(sVt, 48, 32, 64), acc3);
  }
  __syncthreads();
#pragma unroll
  for (int j = 0; j < 8; ++j){
    int m = 16 * w + 8 * g + j;
    float* Tp = wsTacc + bh * LMSZ + m * 64 + n;
    atomicAdd(Tp + 0,  acc0[j]);
    atomicAdd(Tp + 16, acc1[j]);
    atomicAdd(Tp + 32, acc2[j]);
    atomicAdd(Tp + 48, acc3[j]);
  }
  if (t < 64) atomicAdd(&wsRow[bh * 64 + t], sR[t]);
}

// ---------------- Kernel B3: Newton-Schulz inverse + W = inv(K2) @ Tn ----------------
__device__ __forceinline__ void mm64(unsigned short* D, const unsigned short* A,
                                     const unsigned short* Bt){
  int t = threadIdx.x, w = t >> 5, lane = t & 31, g = lane >> 4, n = lane & 15;
  v16bf a0 = ld_frag(A, 16 * w, 0, 64), a1 = ld_frag(A, 16 * w, 32, 64);
#pragma unroll
  for (int nt = 0; nt < 4; ++nt){
    v8f acc = {};
    acc = wmma_bf16(a0, ld_frag(Bt, 16 * nt, 0, 64), acc);
    acc = wmma_bf16(a1, ld_frag(Bt, 16 * nt, 32, 64), acc);
#pragma unroll
    for (int j = 0; j < 8; ++j){
      int m = 16 * w + 8 * g + j;
      D[m * 64 + 16 * nt + n] = f2bf(acc[j]);
    }
  }
}

__global__ void inv_kernel(const float* __restrict__ wsK2, const float* __restrict__ wsGmax,
                           const float* __restrict__ wsTacc, const float* __restrict__ wsRow,
                           float* __restrict__ wsW){
  int bh = blockIdx.x, t = threadIdx.x;
  __shared__ __align__(16) unsigned short KmA[4096], ViA[4096], ViT[4096], KV[4096], R[4096], T0[4096];
  float ig = 1.0f / wsGmax[0];
  for (int i = t; i < 4096; i += 128){
    int m = i >> 6, c = i & 63;
    KmA[i] = f2bf(wsK2[bh * LMSZ + i]);
    float v = wsK2[bh * LMSZ + c * 64 + m] * ig;       // Vi = Km^T / gmax
    ViA[i] = f2bf(v);
    ViT[c * 64 + m] = f2bf(v);
  }
  __syncthreads();
  for (int it = 0; it < 6; ++it){
    mm64(KV, KmA, ViT); __syncthreads();               // KV = Km @ Vi
    for (int i = t; i < 4096; i += 128){ int m = i >> 6, c = i & 63;
      T0[c * 64 + m] = f2bf((m == c ? 7.f : 0.f) - bf2f(KV[i])); }
    __syncthreads();
    mm64(R, KV, T0); __syncthreads();                  // KV @ (7I - KV)
    for (int i = t; i < 4096; i += 128){ int m = i >> 6, c = i & 63;
      T0[c * 64 + m] = f2bf((m == c ? 15.f : 0.f) - bf2f(R[i])); }
    __syncthreads();
    mm64(R, KV, T0); __syncthreads();                  // KV @ (15I - ...)
    for (int i = t; i < 4096; i += 128){ int m = i >> 6, c = i & 63;
      T0[c * 64 + m] = f2bf((m == c ? 13.f : 0.f) - bf2f(R[i])); }
    __syncthreads();
    mm64(R, ViA, T0); __syncthreads();                 // Vi @ (13I - ...)
    for (int i = t; i < 4096; i += 128){ int m = i >> 6, c = i & 63;
      float v = 0.25f * bf2f(R[i]);
      ViA[i] = f2bf(v); ViT[c * 64 + m] = f2bf(v); }
    __syncthreads();
  }
  // Tn (normalized kernel_3 @ V), transposed as B operand
  for (int i = t; i < 4096; i += 128){
    int m = i >> 6, c = i & 63;
    T0[c * 64 + m] = f2bf(wsTacc[bh * LMSZ + i] / wsRow[bh * 64 + m]);
  }
  __syncthreads();
  { // W = Vi @ Tn, stored f32
    int w = t >> 5, lane = t & 31, g = lane >> 4, n = lane & 15;
    v16bf a0 = ld_frag(ViA, 16 * w, 0, 64), a1 = ld_frag(ViA, 16 * w, 32, 64);
#pragma unroll
    for (int nt = 0; nt < 4; ++nt){
      v8f acc = {};
      acc = wmma_bf16(a0, ld_frag(T0, 16 * nt, 0, 64), acc);
      acc = wmma_bf16(a1, ld_frag(T0, 16 * nt, 32, 64), acc);
#pragma unroll
      for (int j = 0; j < 8; ++j){
        int m = 16 * w + 8 * g + j;
        wsW[bh * LMSZ + m * 64 + 16 * nt + n] = acc[j];
      }
    }
  }
}

// ---------------- Kernel D: X = softmax(Qs @ Kl^T) @ W ----------------
// grid (96, 64); block 128 (4 waves), 64 query rows per block.
__global__ void out_kernel(const float* __restrict__ Q, const float* __restrict__ mask,
                           const float* __restrict__ wsKl, const float* __restrict__ wsW,
                           float* __restrict__ out){
  int bh = blockIdx.x, b = bh / NHEAD, qb = blockIdx.y;
  int t = threadIdx.x, w = t >> 5, lane = t & 31, g = lane >> 4, n = lane & 15;
  __shared__ __align__(16) unsigned short sQ[4096], sKl[4096], sWt[4096], sP[4096];
  __shared__ __align__(16) float sF[4096];
  __shared__ float sM[64];
  if (t < 64) sM[t] = mask[b * SEQ + qb * 64 + t];
  __syncthreads();
  {
    const float4* Q4  = (const float4*)(Q + ((size_t)bh * SEQ + (size_t)qb * 64) * HD);
    const float4* Kl4 = (const float4*)(wsKl + bh * LMSZ);
    const float4* W4  = (const float4*)(wsW + bh * LMSZ);
    for (int j = t; j < 1024; j += 128){
      int r = j >> 4, c4 = (j & 15) * 4;
      float4 kl = Kl4[j];
      uint2 u; u.x = pk2(kl.x, kl.y); u.y = pk2(kl.z, kl.w);
      *(uint2*)(&sKl[r * 64 + c4]) = u;
      float4 wv = W4[j];
      sWt[(c4 + 0) * 64 + r] = f2bf(wv.x);              // W transposed (B operand)
      sWt[(c4 + 1) * 64 + r] = f2bf(wv.y);
      sWt[(c4 + 2) * 64 + r] = f2bf(wv.z);
      sWt[(c4 + 3) * 64 + r] = f2bf(wv.w);
      float4 qv = Q4[j];
      float ms = sM[r] * kScale;
      uint2 uq; uq.x = pk2(qv.x * ms, qv.y * ms); uq.y = pk2(qv.z * ms, qv.w * ms);
      *(uint2*)(&sQ[r * 64 + c4]) = uq;
    }
  }
  __syncthreads();
  v16bf a0 = ld_frag(sQ, 16 * w, 0, 64), a1 = ld_frag(sQ, 16 * w, 32, 64);
#pragma unroll
  for (int lt = 0; lt < 4; ++lt){
    v8f acc = {};
    acc = wmma_bf16(a0, ld_frag(sKl, 16 * lt, 0, 64), acc);
    acc = wmma_bf16(a1, ld_frag(sKl, 16 * lt, 32, 64), acc);
#pragma unroll
    for (int j = 0; j < 8; ++j){
      int m = 16 * w + 8 * g + j;
      sF[m * 64 + 16 * lt + n] = acc[j];
    }
  }
  __syncthreads();
  if (t < 64){                                        // exact row softmax over 64 landmarks
    float mx = -1e30f;
    for (int c = 0; c < 64; ++c) mx = fmaxf(mx, sF[t * 64 + c]);
    float sm = 0.f;
    for (int c = 0; c < 64; ++c){ float e = __expf(sF[t * 64 + c] - mx); sF[t * 64 + c] = e; sm += e; }
    float inv = 1.0f / sm;
    for (int c = 0; c < 64; ++c) sP[t * 64 + c] = f2bf(sF[t * 64 + c] * inv);
  }
  __syncthreads();
  v16bf p0 = ld_frag(sP, 16 * w, 0, 64), p1 = ld_frag(sP, 16 * w, 32, 64);
#pragma unroll
  for (int nt = 0; nt < 4; ++nt){
    v8f acc = {};
    acc = wmma_bf16(p0, ld_frag(sWt, 16 * nt, 0, 64), acc);
    acc = wmma_bf16(p1, ld_frag(sWt, 16 * nt, 32, 64), acc);
#pragma unroll
    for (int j = 0; j < 8; ++j){
      int m = 16 * w + 8 * g + j;
      sF[m * 64 + 16 * nt + n] = acc[j];
    }
  }
  __syncthreads();
  float4* op4 = (float4*)(out + ((size_t)bh * SEQ + (size_t)qb * 64) * HD);
  const float4* sF4 = (const float4*)sF;
  for (int j = t; j < 1024; j += 128) op4[j] = sF4[j];   // contiguous b128 stores
}

extern "C" void kernel_launch(void* const* d_in, const int* in_sizes, int n_in,
                              void* d_out, int out_size, void* d_ws, size_t ws_size,
                              hipStream_t stream){
  (void)in_sizes; (void)n_in; (void)out_size; (void)ws_size;
  const float* Q    = (const float*)d_in[0];
  const float* K    = (const float*)d_in[1];
  const float* V    = (const float*)d_in[2];
  const float* mask = (const float*)d_in[3];
  float* ws = (float*)d_ws;
  const size_t M = (size_t)BH * LMSZ;
  float* wsQl   = ws;
  float* wsKl   = ws + M;
  float* wsK2   = ws + 2 * M;
  float* wsTacc = ws + 3 * M;
  float* wsRow  = ws + 4 * M;              // 96*64
  float* wsW    = wsRow + (size_t)BH * 64;
  float* wsCmax = wsW + M;                 // 96
  float* wsGmax = wsCmax + BH;             // 1

  hipMemsetAsync(wsTacc, 0, (M + (size_t)BH * 64) * sizeof(float), stream);
  lm_kernel    <<<dim3(BH, 8),  128, 0, stream>>>(Q, K, mask, wsQl, wsKl);
  k2_kernel    <<<BH,           64,  0, stream>>>(wsQl, wsKl, wsK2, wsCmax);
  gmax_kernel  <<<1,            128, 0, stream>>>(wsCmax, wsGmax);
  flash3_kernel<<<dim3(BH, 8),  128, 0, stream>>>(K, V, mask, wsQl, wsTacc, wsRow);
  inv_kernel   <<<BH,           128, 0, stream>>>(wsK2, wsGmax, wsTacc, wsRow, wsW);
  out_kernel   <<<dim3(BH, 64), 128, 0, stream>>>(Q, mask, wsKl, wsW, (float*)d_out);
}
